// GRANModel_64665027608959
// MI455X (gfx1250) — matrix operand order repaired and checked by reading, not compiled, not run on hardware
//
#include <hip/hip_runtime.h>
#include <hip/hip_fp16.h>

typedef __attribute__((ext_vector_type(16))) _Float16 v16h;
typedef __attribute__((ext_vector_type(8)))  float    v8f;

constexpr int B_    = 1024;
constexpr int S_    = 7;
constexpr int E_    = 256;
constexpr int H_    = 4;
constexpr int L_    = 12;
constexpr int DH_   = E_ / H_;      // 64
constexpr int INNER_= 512;
constexpr int VOC_  = 50000;
constexpr int NREL_ = 200;
constexpr int BS_   = B_ * S_;      // 7168

// ---------------------------------------------------------------- helpers

__device__ __forceinline__ float gelu_exact(float x) {
  return 0.5f * x * (1.0f + erff(x * 0.70710678118654752f));
}

__device__ __forceinline__ float wave_sum(float v) {
  #pragma unroll
  for (int o = 16; o > 0; o >>= 1) v += __shfl_xor(v, o, 32);
  return v;
}

// ---------------------------------------------------------------- convert

__global__ __launch_bounds__(256) void cvt_f16_kernel(
    const float* __restrict__ in, _Float16* __restrict__ out, int n) {
  int i = blockIdx.x * 256 + threadIdx.x;
  if (i < n) out[i] = (_Float16)in[i];
}

// in: [K,N] f32 row-major  ->  out: [N,K] f16 row-major
__global__ __launch_bounds__(256) void cvt_tr_f16_kernel(
    const float* __restrict__ in, _Float16* __restrict__ out, int K, int N) {
  int i = blockIdx.x * 256 + threadIdx.x;
  if (i >= K * N) return;
  int n = i / K;
  int k = i - n * K;
  out[i] = (_Float16)in[(size_t)k * N + n];
}

// ---------------------------------------------------------------- WMMA GEMM
// C[M,N] = A[M,K](f16,row) * Bt[N,K](f16,row)^T + bias[N]
// act: 0=none, 1=exact GELU.  maskType != null => vocab-logit epilogue.
// Wave tile 32x64, block tile 64x256 (8 waves).

__global__ __launch_bounds__(256) void gemm_f16_wmma(
    const _Float16* __restrict__ A, const _Float16* __restrict__ Bt,
    const float* __restrict__ bias, float* __restrict__ C,
    _Float16* __restrict__ C16, const float* __restrict__ maskType,
    int M, int N, int K, int act) {
  const int lane = threadIdx.x & 31;
  const int wave = threadIdx.x >> 5;
  const int wm   = wave >> 2;          // 0..1
  const int wn   = wave & 3;           // 0..3
  const int lr   = lane & 15;
  const int sel  = lane >> 4;          // half-wave select

  const int mBase = blockIdx.y * 64 + wm * 32;
  const int nBase = blockIdx.x * 256 + wn * 64;

  const v8f vz = {0.f, 0.f, 0.f, 0.f, 0.f, 0.f, 0.f, 0.f};
  v8f acc[2][4];
  #pragma unroll
  for (int i = 0; i < 2; ++i)
    #pragma unroll
    for (int j = 0; j < 4; ++j) acc[i][j] = vz;

  // A fragment rows (16x32 f16 layout: lanes 0-15 hold K[0..7]&K[16..23],
  // lanes 16-31 hold K[8..15]&K[24..31] of rows M=0..15)
  const _Float16* aRow0 = A + (size_t)(mBase + lr) * K + 8 * sel;
  const _Float16* aRow1 = A + (size_t)(mBase + 16 + lr) * K + 8 * sel;

  // B fragment columns (32x16: lanes 0-15 = K[0..7]&K[16..23], lanes 16-31 = K[8..15]&K[24..31])
  const _Float16* bRow[4];
  #pragma unroll
  for (int j = 0; j < 4; ++j) {
    int col = nBase + 16 * j + lr;
    if (col >= N) col = N - 1;          // clamp; store is guarded
    bRow[j] = Bt + (size_t)col * K + 8 * sel;
  }

  for (int k0 = 0; k0 < K; k0 += 32) {
    union Frag { v16h v; float4 f4[2]; };
    Frag a0, a1, bf[4];

    const _Float16* p0 = aRow0 + k0;
    a0.f4[0] = *reinterpret_cast<const float4*>(p0);
    a0.f4[1] = *reinterpret_cast<const float4*>(p0 + 16);
    const _Float16* p1 = aRow1 + k0;
    a1.f4[0] = *reinterpret_cast<const float4*>(p1);
    a1.f4[1] = *reinterpret_cast<const float4*>(p1 + 16);

    #pragma unroll
    for (int j = 0; j < 4; ++j) {
      const _Float16* pb = bRow[j] + k0;
      bf[j].f4[0] = *reinterpret_cast<const float4*>(pb);
      bf[j].f4[1] = *reinterpret_cast<const float4*>(pb + 16);
      // Unconditional speculative prefetch of next K-step (safe: dropped on
      // translation failure; keeps the inner loop branch-free).
      __builtin_prefetch(pb + 32, 0, 3);
    }

    #pragma unroll
    for (int j = 0; j < 4; ++j) {
      acc[0][j] = __builtin_amdgcn_wmma_f32_16x16x32_f16(
          false, a0.v, false, bf[j].v, (short)0, acc[0][j], false, false);
      acc[1][j] = __builtin_amdgcn_wmma_f32_16x16x32_f16(
          false, a1.v, false, bf[j].v, (short)0, acc[1][j], false, false);
    }
  }

  // Epilogue. C/D layout: VGPR r, lanes 0-15 -> (M=r,  N=lane),
  //                               lanes 16-31 -> (M=8+r,N=lane-16)
  #pragma unroll
  for (int i = 0; i < 2; ++i) {
    const int rowB = mBase + 16 * i + 8 * sel;
    #pragma unroll
    for (int j = 0; j < 4; ++j) {
      const int col = nBase + 16 * j + lr;
      if (col >= N) continue;
      const float bv = bias ? bias[col] : 0.f;
      #pragma unroll
      for (int r = 0; r < 8; ++r) {
        const int row = rowB + r;
        float v = acc[i][j][r] + bv;
        if (act == 1) v = gelu_exact(v);
        if (maskType) {
          float mt = maskType[row];
          float ti = (col < 2) ? 0.f
                   : (col < 2 + NREL_) ? ((mt < 0.f) ? 1.f : 0.f)
                                       : ((mt > 0.f) ? 1.f : 0.f);
          v += 1e6f * (ti - 1.f);
        }
        C[(size_t)row * N + col] = v;
        if (C16) C16[(size_t)row * N + col] = (_Float16)v;
      }
    }
  }
}

// ---------------------------------------------------------------- embed + LN

__global__ __launch_bounds__(256) void embed_ln_kernel(
    const int* __restrict__ ids, const float* __restrict__ emb,
    const float* __restrict__ g, const float* __restrict__ beta,
    float* __restrict__ xo, _Float16* __restrict__ x16) {
  const int tok = blockIdx.x * 8 + (threadIdx.x >> 5);
  if (tok >= BS_) return;
  const int lane = threadIdx.x & 31;
  const float* row = emb + (size_t)ids[tok] * E_;
  float vl[8], s = 0.f, sq = 0.f;
  #pragma unroll
  for (int i = 0; i < 8; ++i) {
    float x = row[lane + 32 * i];
    vl[i] = x; s += x; sq += x * x;
  }
  s = wave_sum(s); sq = wave_sum(sq);
  const float mu = s * (1.f / E_);
  const float rs = rsqrtf(sq * (1.f / E_) - mu * mu + 1e-12f);
  #pragma unroll
  for (int i = 0; i < 8; ++i) {
    const int c = lane + 32 * i;
    float y = (vl[i] - mu) * rs * g[c] + beta[c];
    xo[(size_t)tok * E_ + c]  = y;
    x16[(size_t)tok * E_ + c] = (_Float16)y;
  }
}

// x = LN(x + delta)  (in place on f32, mirror f16)
__global__ __launch_bounds__(256) void resid_ln_kernel(
    float* __restrict__ x, const float* __restrict__ delta,
    const float* __restrict__ g, const float* __restrict__ beta,
    _Float16* __restrict__ x16, float eps) {
  const int tok = blockIdx.x * 8 + (threadIdx.x >> 5);
  if (tok >= BS_) return;
  const int lane = threadIdx.x & 31;
  float vl[8], s = 0.f, sq = 0.f;
  #pragma unroll
  for (int i = 0; i < 8; ++i) {
    const int c = lane + 32 * i;
    float v = x[(size_t)tok * E_ + c] + delta[(size_t)tok * E_ + c];
    vl[i] = v; s += v; sq += v * v;
  }
  s = wave_sum(s); sq = wave_sum(sq);
  const float mu = s * (1.f / E_);
  const float rs = rsqrtf(sq * (1.f / E_) - mu * mu + eps);
  #pragma unroll
  for (int i = 0; i < 8; ++i) {
    const int c = lane + 32 * i;
    float y = (vl[i] - mu) * rs * g[c] + beta[c];
    x[(size_t)tok * E_ + c]  = y;
    x16[(size_t)tok * E_ + c] = (_Float16)y;
  }
}

// ---------------------------------------------------------------- edge bias

__global__ __launch_bounds__(256) void edge_bias_kernel(
    const int* __restrict__ lbl, const float* __restrict__ ekw,
    const float* __restrict__ evw, float* __restrict__ ek,
    float* __restrict__ ev) {
  const int i = blockIdx.x * 256 + threadIdx.x;
  if (i >= S_ * S_ * DH_) return;
  const int p = i / DH_, d = i - p * DH_;
  const int l = lbl[p];
  const float m = (l > 0) ? 1.f : 0.f;
  ek[i] = ekw[l * DH_ + d] * m;
  ev[i] = evw[l * DH_ + d] * m;
}

// ---------------------------------------------------------------- attention
// one thread per (b,h,qi); S=7, DH=64 -> negligible FLOPs, plain VALU.

__global__ __launch_bounds__(256) void attn_kernel(
    const float* __restrict__ q, const float* __restrict__ k,
    const float* __restrict__ v, const float* __restrict__ ek,
    const float* __restrict__ ev, const float* __restrict__ mask,
    _Float16* __restrict__ ctx) {
  const int tid = blockIdx.x * blockDim.x + threadIdx.x;
  if (tid >= B_ * H_ * S_) return;
  const int b  = tid / (H_ * S_);
  const int r  = tid - b * (H_ * S_);
  const int h  = r / S_;
  const int qi = r - h * S_;

  const float* qp = q + ((size_t)(b * S_ + qi) * H_ + h) * DH_;
  float qr[DH_];
  #pragma unroll
  for (int d = 0; d < DH_; ++d) qr[d] = qp[d];

  const float mq = mask[b * S_ + qi];
  float sc[S_], mx = -1e30f;
  for (int j = 0; j < S_; ++j) {
    const float* kp = k + ((size_t)(b * S_ + j) * H_ + h) * DH_;
    const float* ep = ek + (size_t)(qi * S_ + j) * DH_;
    float s = 0.f;
    for (int d = 0; d < DH_; ++d) s += qr[d] * (kp[d] + ep[d]);
    s = s * 0.125f + 1e6f * (mq * mask[b * S_ + j] - 1.f);
    sc[j] = s;
    mx = fmaxf(mx, s);
  }
  float z = 0.f;
  #pragma unroll
  for (int j = 0; j < S_; ++j) { sc[j] = expf(sc[j] - mx); z += sc[j]; }
  const float iz = 1.f / z;

  _Float16* op = ctx + ((size_t)(b * S_ + qi) * H_ + h) * DH_;
  for (int d = 0; d < DH_; ++d) {
    float a = 0.f;
    #pragma unroll
    for (int j = 0; j < S_; ++j) {
      a += sc[j] * (v[((size_t)(b * S_ + j) * H_ + h) * DH_ + d] +
                    ev[(size_t)(qi * S_ + j) * DH_ + d]);
    }
    op[d] = (_Float16)(a * iz);
  }
}

// ---------------------------------------------------------------- final gather

__global__ __launch_bounds__(256) void final_gather_kernel(
    const float* __restrict__ x, const int* __restrict__ pos,
    const float* __restrict__ g, const float* __restrict__ beta,
    _Float16* __restrict__ hm16) {
  const int b = blockIdx.x * 8 + (threadIdx.x >> 5);
  if (b >= B_) return;
  const int lane = threadIdx.x & 31;
  const float* row = x + (size_t)(b * S_ + pos[b]) * E_;
  float vl[8], s = 0.f, sq = 0.f;
  #pragma unroll
  for (int i = 0; i < 8; ++i) {
    float v = gelu_exact(row[lane + 32 * i]);
    vl[i] = v; s += v; sq += v * v;
  }
  s = wave_sum(s); sq = wave_sum(sq);
  const float mu = s * (1.f / E_);
  const float rs = rsqrtf(sq * (1.f / E_) - mu * mu + 1e-7f);
  #pragma unroll
  for (int i = 0; i < 8; ++i) {
    const int c = lane + 32 * i;
    hm16[(size_t)b * E_ + c] = (_Float16)((vl[i] - mu) * rs * g[c] + beta[c]);
  }
}

// ---------------------------------------------------------------- loss

__global__ __launch_bounds__(256) void loss_rows_kernel(
    const float* __restrict__ fc, const int* __restrict__ labels,
    const float* __restrict__ maskType, float* __restrict__ rowloss) {
  __shared__ float sh1[256];
  __shared__ float sh2[256];
  const int b = blockIdx.x;
  const int t = threadIdx.x;
  const float* row = fc + (size_t)b * VOC_;
  const float mt = maskType[b];
  const bool ent = (mt > 0.f);

  float lmax = -1e30f;
  for (int v = t; v < VOC_; v += 256) lmax = fmaxf(lmax, row[v]);
  sh1[t] = lmax; __syncthreads();
  for (int o = 128; o > 0; o >>= 1) {
    if (t < o) sh1[t] = fmaxf(sh1[t], sh1[t + o]);
    __syncthreads();
  }
  const float M = sh1[0];
  __syncthreads();

  float z = 0.f, scand = 0.f;
  for (int v = t; v < VOC_; v += 256) {
    float xv = row[v];
    z += expf(xv - M);
    bool cand = (v >= 2) && (ent ? (v >= 2 + NREL_) : (v < 2 + NREL_));
    if (cand) scand += xv;
  }
  sh1[t] = z; sh2[t] = scand; __syncthreads();
  for (int o = 128; o > 0; o >>= 1) {
    if (t < o) { sh1[t] += sh1[t + o]; sh2[t] += sh2[t + o]; }
    __syncthreads();
  }
  if (t == 0) {
    const float logZ = logf(sh1[0]);
    const float C = ent ? (float)(VOC_ - NREL_ - 2) : (float)NREL_;
    const float lpl = row[labels[b]] - M - logZ;
    const float T = (sh2[0] - C * (M + logZ)) - lpl;   // sum(ti' * logprob)
    const float ncand = C - 1.f;                        // sum(ti) - 1
    rowloss[b] = -(0.9f * lpl + 0.1f * T / ncand);
  }
}

__global__ __launch_bounds__(256) void loss_reduce_kernel(
    const float* __restrict__ rowloss, float* __restrict__ out) {
  __shared__ float sh[256];
  float s = 0.f;
  for (int i = threadIdx.x; i < B_; i += 256) s += rowloss[i];
  sh[threadIdx.x] = s; __syncthreads();
  for (int o = 128; o > 0; o >>= 1) {
    if (threadIdx.x < o) sh[threadIdx.x] += sh[threadIdx.x + o];
    __syncthreads();
  }
  if (threadIdx.x == 0) out[0] = sh[0] * (1.f / B_);
}

// ---------------------------------------------------------------- launcher

static inline int cdiv(int a, int b) { return (a + b - 1) / b; }

extern "C" void kernel_launch(void* const* d_in, const int* in_sizes, int n_in,
                              void* d_out, int out_size, void* d_ws, size_t ws_size,
                              hipStream_t stream) {
  (void)in_sizes; (void)n_in; (void)out_size; (void)ws_size;

  const int*   input_ids  = (const int*)  d_in[0];
  const float* input_mask = (const float*)d_in[1];
  const int*   edge_lbls  = (const int*)  d_in[2];
  const int*   mask_pos   = (const int*)  d_in[3];
  const int*   mask_label = (const int*)  d_in[4];
  const float* mask_type  = (const float*)d_in[5];
  const float* node_emb   = (const float*)d_in[6];
  const float* ln1_g = (const float*)d_in[7];
  const float* ln1_b = (const float*)d_in[8];
  const float* ek_w  = (const float*)d_in[9];
  const float* ev_w  = (const float*)d_in[10];
  const float* Wq = (const float*)d_in[11]; const float* bq = (const float*)d_in[12];
  const float* Wk = (const float*)d_in[13]; const float* bk = (const float*)d_in[14];
  const float* Wv = (const float*)d_in[15]; const float* bv = (const float*)d_in[16];
  const float* Wo = (const float*)d_in[17]; const float* bo = (const float*)d_in[18];
  const float* lna_g = (const float*)d_in[19]; const float* lna_b = (const float*)d_in[20];
  const float* W1 = (const float*)d_in[21]; const float* b1 = (const float*)d_in[22];
  const float* W2 = (const float*)d_in[23]; const float* b2 = (const float*)d_in[24];
  const float* lnf_g = (const float*)d_in[25]; const float* lnf_b = (const float*)d_in[26];
  const float* ln2_g = (const float*)d_in[27]; const float* ln2_b = (const float*)d_in[28];
  const float* fc2_bias = (const float*)d_in[29];

  // ---- workspace carve-up (all 256B aligned) ----
  char* p = (char*)d_ws;
  auto take = [&](size_t bytes) -> void* {
    void* r = (void*)p;
    p += (bytes + 255) & ~(size_t)255;
    return r;
  };
  _Float16* nodeF16 = (_Float16*)take((size_t)VOC_ * E_ * 2);       // 25.6 MB
  _Float16* WqT = (_Float16*)take((size_t)L_ * E_ * E_ * 2);
  _Float16* WkT = (_Float16*)take((size_t)L_ * E_ * E_ * 2);
  _Float16* WvT = (_Float16*)take((size_t)L_ * E_ * E_ * 2);
  _Float16* WoT = (_Float16*)take((size_t)L_ * E_ * E_ * 2);
  _Float16* W1T = (_Float16*)take((size_t)L_ * INNER_ * E_ * 2);
  _Float16* W2T = (_Float16*)take((size_t)L_ * E_ * INNER_ * 2);
  float*    xf32 = (float*)   take((size_t)BS_ * E_ * 4);
  _Float16* xf16 = (_Float16*)take((size_t)BS_ * E_ * 2);
  float*    qb   = (float*)   take((size_t)BS_ * E_ * 4);
  float*    kb   = (float*)   take((size_t)BS_ * E_ * 4);
  float*    vb   = (float*)   take((size_t)BS_ * E_ * 4);
  _Float16* act16 = (_Float16*)take((size_t)BS_ * INNER_ * 2);      // ctx16 / h16
  float*    tmp  = (float*)   take((size_t)BS_ * E_ * 4);           // attn_out / ffn_out
  float*    hf32 = (float*)   take((size_t)BS_ * INNER_ * 4);
  float*    ekb  = (float*)   take((size_t)S_ * S_ * DH_ * 4);
  float*    evb  = (float*)   take((size_t)S_ * S_ * DH_ * 4);
  _Float16* hm16 = (_Float16*)take((size_t)B_ * E_ * 2);
  float*    rowloss = (float*)take((size_t)B_ * 4);

  float* out = (float*)d_out;           // out[0]=loss, out[1..]=fc_out [B,VOC]
  float* fc  = out + 1;

  // ---- weight / embedding conversion (f32 -> f16, transposed) ----
  cvt_f16_kernel<<<cdiv(VOC_ * E_, 256), 256, 0, stream>>>(node_emb, nodeF16, VOC_ * E_);
  for (int l = 0; l < L_; ++l) {
    const size_t oE = (size_t)l * E_ * E_;
    const size_t oI = (size_t)l * E_ * INNER_;
    cvt_tr_f16_kernel<<<cdiv(E_ * E_, 256), 256, 0, stream>>>(Wq + oE, WqT + oE, E_, E_);
    cvt_tr_f16_kernel<<<cdiv(E_ * E_, 256), 256, 0, stream>>>(Wk + oE, WkT + oE, E_, E_);
    cvt_tr_f16_kernel<<<cdiv(E_ * E_, 256), 256, 0, stream>>>(Wv + oE, WvT + oE, E_, E_);
    cvt_tr_f16_kernel<<<cdiv(E_ * E_, 256), 256, 0, stream>>>(Wo + oE, WoT + oE, E_, E_);
    cvt_tr_f16_kernel<<<cdiv(E_ * INNER_, 256), 256, 0, stream>>>(W1 + oI, W1T + oI, E_, INNER_);
    cvt_tr_f16_kernel<<<cdiv(E_ * INNER_, 256), 256, 0, stream>>>(W2 + oI, W2T + oI, INNER_, E_);
  }

  // ---- edge K/V bias + input embedding LN ----
  edge_bias_kernel<<<cdiv(S_ * S_ * DH_, 256), 256, 0, stream>>>(edge_lbls, ek_w, ev_w, ekb, evb);
  embed_ln_kernel<<<BS_ / 8, 256, 0, stream>>>(input_ids, node_emb, ln1_g, ln1_b, xf32, xf16);

  const dim3 gEE(cdiv(E_, 256), cdiv(BS_, 64));
  const dim3 gEI(cdiv(INNER_, 256), cdiv(BS_, 64));

  for (int l = 0; l < L_; ++l) {
    const size_t oE = (size_t)l * E_ * E_;
    const size_t oI = (size_t)l * E_ * INNER_;
    // Q, K, V projections
    gemm_f16_wmma<<<gEE, 256, 0, stream>>>(xf16, WqT + oE, bq + l * E_, qb, nullptr, nullptr, BS_, E_, E_, 0);
    gemm_f16_wmma<<<gEE, 256, 0, stream>>>(xf16, WkT + oE, bk + l * E_, kb, nullptr, nullptr, BS_, E_, E_, 0);
    gemm_f16_wmma<<<gEE, 256, 0, stream>>>(xf16, WvT + oE, bv + l * E_, vb, nullptr, nullptr, BS_, E_, E_, 0);
    // relational attention (small: S=7)
    attn_kernel<<<(B_ * H_ * S_) / 256, 256, 0, stream>>>(qb, kb, vb, ekb, evb, input_mask, act16);
    // output projection + residual LN
    gemm_f16_wmma<<<gEE, 256, 0, stream>>>(act16, WoT + oE, bo + l * E_, tmp, nullptr, nullptr, BS_, E_, E_, 0);
    resid_ln_kernel<<<BS_ / 8, 256, 0, stream>>>(xf32, tmp, lna_g + l * E_, lna_b + l * E_, xf16, 1e-12f);
    // FFN: gelu(x@W1+b1) @ W2 + b2, residual LN
    gemm_f16_wmma<<<gEI, 256, 0, stream>>>(xf16, W1T + oI, b1 + l * INNER_, hf32, act16, nullptr, BS_, INNER_, E_, 1);
    gemm_f16_wmma<<<gEE, 256, 0, stream>>>(act16, W2T + oI, b2 + l * E_, tmp, nullptr, nullptr, BS_, E_, INNER_, 0);
    resid_ln_kernel<<<BS_ / 8, 256, 0, stream>>>(xf32, tmp, lnf_g + l * E_, lnf_b + l * E_, xf16, 1e-12f);
  }

  // ---- masked-position gather -> gelu -> LN ----
  final_gather_kernel<<<B_ / 8, 256, 0, stream>>>(xf32, mask_pos, ln2_g, ln2_b, hm16);

  // ---- vocab logits: hm[B,E] @ node_emb[VOC,E]^T + fc2_bias, + type mask ----
  const dim3 gV(cdiv(VOC_, 256), cdiv(B_, 64));
  gemm_f16_wmma<<<gV, 256, 0, stream>>>(hm16, nodeF16, fc2_bias, fc, nullptr, mask_type, B_, VOC_, E_, 0);

  // ---- soft-label cross-entropy loss ----
  loss_rows_kernel<<<B_, 256, 0, stream>>>(fc, mask_label, mask_type, rowloss);
  loss_reduce_kernel<<<1, 256, 0, stream>>>(rowloss, out);
}
